// SimpleRetention_54348516163671
// MI455X (gfx1250) — matrix operand order repaired and checked by reading, not compiled, not run on hardware
//
#include <hip/hip_runtime.h>

#define Hdim 512
#define Bdim 8
#define Slen 2048
#define BS (Bdim * Slen)

typedef __bf16 bf16;
typedef __attribute__((ext_vector_type(16))) __bf16 v16bf;
typedef __attribute__((ext_vector_type(8)))  __bf16 v8bf;
typedef __attribute__((ext_vector_type(8)))  float   v8f;
typedef __attribute__((ext_vector_type(8)))  float   f8;

__device__ __forceinline__ v16bf join8(v8bf lo, v8bf hi) {
  v16bf r;
#pragma unroll
  for (int i = 0; i < 8; i++) { r[i] = lo[i]; r[i + 8] = hi[i]; }
  return r;
}

__device__ __forceinline__ v8f wmma_bf16(v16bf a, v16bf b, v8f c) {
  // D = A(16x32 bf16) * B(32x16 bf16) + C(16x16 f32)
  return __builtin_amdgcn_wmma_f32_16x16x32_bf16(
      /*neg_a=*/false, a, /*neg_b=*/false, b,
      /*c_mod=*/(short)0, c, /*reuse_a=*/false, /*reuse_b=*/false);
}

__device__ __forceinline__ float fast_exp2(float x) {
  return __builtin_amdgcn_exp2f(x);              // v_exp_f32 (TRANS)
}

// ---------------------------------------------------------------------------
// Kernel 0: convert + transpose weights to bf16 (Wt[n][k] = W[k][n])
// ---------------------------------------------------------------------------
__global__ void retn_wt_kernel(const float* __restrict__ WQ,
                               const float* __restrict__ WK,
                               const float* __restrict__ WV,
                               bf16* __restrict__ WtQ,
                               bf16* __restrict__ WtK,
                               bf16* __restrict__ WtV) {
  int idx = blockIdx.x * 256 + threadIdx.x;          // 0 .. H*H-1
  const float* W = (blockIdx.y == 0) ? WQ : (blockIdx.y == 1) ? WK : WV;
  bf16* Wt       = (blockIdx.y == 0) ? WtQ : (blockIdx.y == 1) ? WtK : WtV;
  int n = idx / Hdim, k = idx % Hdim;
  Wt[idx] = (bf16)W[(size_t)k * Hdim + n];
}

// ---------------------------------------------------------------------------
// Kernel 1: Q/K/V projections via bf16 WMMA.
// grid = (BS/128, H/128, 3); block = 256 (8 waves); wave -> 16 rows x 128 cols
// Q,K stored row-major bf16; V stored transposed [b][h][s] bf16.
// ---------------------------------------------------------------------------
__global__ void __launch_bounds__(256)
retn_proj_kernel(const float* __restrict__ X,
                 const bf16* __restrict__ WtQ, const bf16* __restrict__ WtK,
                 const bf16* __restrict__ WtV,
                 bf16* __restrict__ Qb, bf16* __restrict__ Kb,
                 bf16* __restrict__ Vt) {
  int tid = threadIdx.x, wave = tid >> 5, lane = tid & 31;
  int lhalf = lane >> 4, l15 = lane & 15;
  int rowBase = blockIdx.x * 128 + wave * 16;
  int colBase = blockIdx.y * 128;
  int which   = blockIdx.z;
  const bf16* Wt = (which == 0) ? WtQ : (which == 1) ? WtK : WtV;

  v8f acc[8];
#pragma unroll
  for (int j = 0; j < 8; j++) acc[j] = (v8f){0.f, 0.f, 0.f, 0.f, 0.f, 0.f, 0.f, 0.f};

  const float* xrow = X + (size_t)(rowBase + l15) * Hdim;
  const bf16*  wcol = Wt + (size_t)(colBase + l15) * Hdim + lhalf * 16;

  // software pipeline the A-chunk loads across c
  f8 xa = *(const f8*)(xrow + lhalf * 16);
  f8 xb = *(const f8*)(xrow + lhalf * 16 + 8);

  for (int c = 0; c < 16; c++) {               // K = 512 in chunks of 32
    v16bf a;
#pragma unroll
    for (int i = 0; i < 8; i++) { a[i] = (bf16)xa[i]; a[i + 8] = (bf16)xb[i]; }
    if (c < 15) {                              // prefetch next A chunk
      int off1 = (c + 1) * 32 + lhalf * 16;
      xa = *(const f8*)(xrow + off1);
      xb = *(const f8*)(xrow + off1 + 8);
    }
    // 4-deep B-tile double buffering: 4 loads in flight per WMMA drain
    v16bf bm[4];
#pragma unroll
    for (int jj = 0; jj < 8; jj += 4) {
#pragma unroll
      for (int u = 0; u < 4; u++)
        bm[u] = *(const v16bf*)(wcol + (size_t)(jj + u) * 16 * Hdim + c * 32);
#pragma unroll
      for (int u = 0; u < 4; u++)
        acc[jj + u] = wmma_bf16(a, bm[u], acc[jj + u]);
    }
  }

  if (which < 2) {                             // Q, K row-major bf16
    bf16* Out = (which == 0) ? Qb : Kb;
#pragma unroll
    for (int j = 0; j < 8; j++) {
      int col = colBase + j * 16 + l15;
#pragma unroll
      for (int r = 0; r < 8; r++) {
        int row = rowBase + r + 8 * lhalf;     // C layout: M = r + 8*(lane/16)
        Out[(size_t)row * Hdim + col] = (bf16)acc[j][r];
      }
    }
  } else {                                     // V transposed: Vt[b][h][s]
    int b = rowBase / Slen;
    int sBase = rowBase % Slen;
#pragma unroll
    for (int j = 0; j < 8; j++) {
      int h = colBase + j * 16 + l15;
      v8bf pk;
#pragma unroll
      for (int r = 0; r < 8; r++) pk[r] = (bf16)acc[j][r];
      *(v8bf*)(Vt + (size_t)b * Hdim * Slen + (size_t)h * Slen + sBase + 8 * lhalf) = pk;
    }
  }
}

// ---------------------------------------------------------------------------
// Kernel 2: fused retention  out = swish((Q K^T) * D) V
// grid = (BS/64); block = 256 (8 waves = 4 row-groups x 2 hidden-halves)
// ---------------------------------------------------------------------------
__global__ void __launch_bounds__(256)
retn_core_kernel(const bf16* __restrict__ Qb, const bf16* __restrict__ Kb,
                 const bf16* __restrict__ Vt, float* __restrict__ Out) {
  __shared__ __align__(16) bf16 Pst[4][16][40];   // 16x32 P per group, padded rows

  int tid = threadIdx.x, wave = tid >> 5, lane = tid & 31;
  int lhalf = lane >> 4, l15 = lane & 15;
  int g = wave >> 1, hh = wave & 1;
  int rowBase = blockIdx.x * 64 + g * 16;
  int b    = rowBase / Slen;
  int sRow = rowBase % Slen;
  int hBase = hh * 256;

  // Q rows register-resident in A-operand layout (16 chunks of K=32).
  v16bf qa[16];
  const bf16* qrow = Qb + (size_t)(rowBase + l15) * Hdim;
#pragma unroll
  for (int c = 0; c < 16; c++) {
    int off1 = c * 32 + lhalf * 16;
    qa[c] = join8(*(const v8bf*)(qrow + off1), *(const v8bf*)(qrow + off1 + 8));
  }

  v8f acc[16];
#pragma unroll
  for (int j = 0; j < 16; j++) acc[j] = (v8f){0.f, 0.f, 0.f, 0.f, 0.f, 0.f, 0.f, 0.f};

  const float LOG2G = -0.04580368961312478f;      // log2(0.96875)
  const float LOG2E = 1.4426950408889634f;        // log2(e)
  const bf16* Kbb = Kb + (size_t)b * Slen * Hdim;
  const bf16* Vtb = Vt + (size_t)b * Hdim * Slen;
  const bf16* vcol = Vtb + (size_t)(hBase + l15) * Slen + lhalf * 16;

  for (int t = 0; t < Slen; t += 32) {
    int scolBase = t + hh * 16;
    const bf16* krow = Kbb + (size_t)(scolBase + l15) * Hdim + lhalf * 16;
    if (t + 32 < Slen) {                        // warm L2/WGP$ one step ahead
      __builtin_prefetch((const void*)(krow + (size_t)32 * Hdim), 0, 1);
      __builtin_prefetch((const void*)(vcol + t + 32), 0, 1);
    }
    // ---- scores: 16x16 tile (seq cols scolBase..+15), K = 512.
    // Two independent accumulation chains + 4-deep load buffering.
    v8f sc0 = (v8f){0.f, 0.f, 0.f, 0.f, 0.f, 0.f, 0.f, 0.f};
    v8f sc1 = (v8f){0.f, 0.f, 0.f, 0.f, 0.f, 0.f, 0.f, 0.f};
    v16bf kbuf[4];
#pragma unroll
    for (int cc = 0; cc < 16; cc += 4) {
#pragma unroll
      for (int u = 0; u < 4; u++)
        kbuf[u] = *(const v16bf*)(krow + (cc + u) * 32);
      sc0 = wmma_bf16(qa[cc + 0], kbuf[0], sc0);
      sc1 = wmma_bf16(qa[cc + 1], kbuf[1], sc1);
      sc0 = wmma_bf16(qa[cc + 2], kbuf[2], sc0);
      sc1 = wmma_bf16(qa[cc + 3], kbuf[3], sc1);
    }
    v8f sc = sc0 + sc1;
    // ---- decay * swish, write P tile (bf16) to LDS
    int scol = scolBase + l15;
#pragma unroll
    for (int r = 0; r < 8; r++) {
      int srow = sRow + r + 8 * lhalf;
      float d = fast_exp2(LOG2G * __builtin_fabsf((float)(srow - scol)));
      float x = sc[r] * d;
      float p = x / (1.f + fast_exp2(-x * LOG2E));   // swish(x) = x*sigmoid(x)
      Pst[g][r + 8 * lhalf][hh * 16 + l15] = (bf16)p;
    }
    __syncthreads();
    // ---- reread P(16x32) in A-operand layout, multiply by V tile
    {
      const bf16* pr = &Pst[g][l15][0];
      int off1 = lhalf * 16;
      v16bf pa = join8(*(const v8bf*)(pr + off1), *(const v8bf*)(pr + off1 + 8));
      const bf16* vrow = vcol + t;
      v16bf vbuf[4];
#pragma unroll
      for (int jj = 0; jj < 16; jj += 4) {
#pragma unroll
        for (int u = 0; u < 4; u++)
          vbuf[u] = *(const v16bf*)(vrow + (size_t)(jj + u) * 16 * Slen);
#pragma unroll
        for (int u = 0; u < 4; u++)
          acc[jj + u] = wmma_bf16(pa, vbuf[u], acc[jj + u]);
      }
    }
    __syncthreads();                             // protect Pst before next step
  }

  // ---- store f32 output [B*S, H]
#pragma unroll
  for (int j = 0; j < 16; j++) {
    int col = hBase + j * 16 + l15;
#pragma unroll
    for (int r = 0; r < 8; r++) {
      int row = rowBase + r + 8 * lhalf;
      Out[(size_t)row * Hdim + col] = acc[j][r];
    }
  }
}

// ---------------------------------------------------------------------------
extern "C" void kernel_launch(void* const* d_in, const int* in_sizes, int n_in,
                              void* d_out, int out_size, void* d_ws, size_t ws_size,
                              hipStream_t stream) {
  const float* X  = (const float*)d_in[0];
  const float* WQ = (const float*)d_in[1];
  const float* WK = (const float*)d_in[2];
  const float* WV = (const float*)d_in[3];
  float* out = (float*)d_out;

  bf16* Qb  = (bf16*)d_ws;
  bf16* Kb  = Qb + (size_t)BS * Hdim;
  bf16* Vt  = Kb + (size_t)BS * Hdim;
  bf16* WtQ = Vt + (size_t)BS * Hdim;
  bf16* WtK = WtQ + (size_t)Hdim * Hdim;
  bf16* WtV = WtK + (size_t)Hdim * Hdim;

  retn_wt_kernel<<<dim3(Hdim * Hdim / 256, 3), 256, 0, stream>>>(WQ, WK, WV, WtQ, WtK, WtV);
  retn_proj_kernel<<<dim3(BS / 128, Hdim / 128, 3), 256, 0, stream>>>(X, WtQ, WtK, WtV, Qb, Kb, Vt);
  retn_core_kernel<<<dim3(BS / 64), 256, 0, stream>>>(Qb, Kb, Vt, out);
}
